// AutoInt_62156766707848
// MI455X (gfx1250) — matrix operand order repaired and checked by reading, not compiled, not run on hardware
//
#include <hip/hip_runtime.h>
#include <hip/hip_bf16.h>
#include <stdint.h>

// AutoInt fused forward for MI455X (gfx1250, wave32, WMMA f16->f32, async-to-LDS).
// B=2048 F=64 D=128 H=8 P=64 V=100000.

#define Bn   2048
#define Fn   64
#define Dn   128
#define Hn   8
#define Pn   64
#define HPn  (Hn * Pn)   // 512
#define Vn   100000

typedef _Float16 v16h __attribute__((ext_vector_type(16)));
typedef _Float16 v8h  __attribute__((ext_vector_type(8)));
typedef _Float16 v4h  __attribute__((ext_vector_type(4)));
typedef float    v8f  __attribute__((ext_vector_type(8)));
typedef float    v4f  __attribute__((ext_vector_type(4)));

__device__ __forceinline__ v8f wmma_f16(v16h a, v16h b, v8f c) {
  return __builtin_amdgcn_wmma_f32_16x16x32_f16(
      /*neg_a=*/false, a, /*neg_b=*/false, b,
      /*c_mod=*/(short)0, c, /*reuse_a=*/false, /*reuse_b=*/false);
}

// ---- fragment builders (CDNA5 16x16x32 f16 VGPR layouts, wave32) ----------

// A fragment (16x32, row-major source in LDS): per lane two contiguous 16B loads.
__device__ __forceinline__ v16h afrag_lds(const _Float16* base, int m0, int k0, int ld) {
  const int lane = threadIdx.x & 31;
  const int half = lane >> 4;
  const int m    = lane & 15;
  const _Float16* p = base + (m0 + m) * ld + k0 + 16 * half;
  const v8h lo = *(const v8h*)p;          // K = k0 + 16*half + 0..7   (VGPR 0..3)
  const v8h hi = *(const v8h*)(p + 8);    // K = k0 + 16*half + 8..15 (VGPR 4..7)
  v16h a;
#pragma unroll
  for (int i = 0; i < 8; ++i) { a[i] = lo[i]; a[8 + i] = hi[i]; }
  return a;
}

// B fragment (32x16) from a source holding B^T row-major:
// element (K,N) = base[(n0+N)*ld + k0+K]  -> one contiguous 32B load per lane.
__device__ __forceinline__ v16h bfrag_lds_nk(const _Float16* base, int k0, int n0, int ld) {
  const int lane = threadIdx.x & 31;
  const int half = lane >> 4;
  const int n    = lane & 15;
  return *(const v16h*)(base + (n0 + n) * ld + k0 + 16 * half);
}

// C fragment (16x16 f32) stores; element (M,N): M = v + 8*half, N = lane&15
__device__ __forceinline__ void cstore_f16(_Float16* base, int m0, int n0, int ld, v8f c) {
  const int lane = threadIdx.x & 31;
  const int half = lane >> 4;
  const int n    = lane & 15;
#pragma unroll
  for (int v = 0; v < 8; ++v)
    base[(m0 + v + 8 * half) * ld + (n0 + n)] = (_Float16)c[v];
}

// Transposed C store: dest[(n0+N)*ld + m0+M] -> 8 contiguous halves = 1 ds_store_b128
__device__ __forceinline__ void cstore_f16_T(_Float16* base, int m0, int n0, int ld, v8f c) {
  const int lane = threadIdx.x & 31;
  const int half = lane >> 4;
  const int n    = lane & 15;
  v8h t;
#pragma unroll
  for (int v = 0; v < 8; ++v) t[v] = (_Float16)c[v];
  *(v8h*)(base + (n0 + n) * ld + m0 + 8 * half) = t;
}

__device__ __forceinline__ void cstore_f32(float* base, int m0, int n0, int ld, v8f c) {
  const int lane = threadIdx.x & 31;
  const int half = lane >> 4;
  const int n    = lane & 15;
#pragma unroll
  for (int v = 0; v < 8; ++v)
    base[(m0 + v + 8 * half) * ld + (n0 + n)] = c[v];
}

// ---------------------------------------------------------------------------
// Prep 0: convert embedding table to f16 (one-time; halves gather traffic and
// enables direct async global->LDS copies with no converts in the main kernel).
__global__ __launch_bounds__(256)
void convert_table(const float* __restrict__ emb_table, _Float16* __restrict__ embF16) {
  const size_t gid  = (size_t)blockIdx.x * blockDim.x + threadIdx.x;
  const size_t base = gid * 16;
  if (base >= (size_t)Vn * Dn) return;
#pragma unroll
  for (int j = 0; j < 4; ++j) {
    const v4f x = *(const v4f*)(emb_table + base + j * 4);
    v4h y;
#pragma unroll
    for (int i = 0; i < 4; ++i) y[i] = (_Float16)x[i];
    *(v4h*)(embF16 + base + j * 4) = y;
  }
}

// Prep 1: pack the four projection weights into per-fragment f16 layout.
// Layout: frag = ((proj*H + h)*4 + nt)*4 + kk ; each frag = 32 lanes * 16 halves.
__global__ __launch_bounds__(256)
void pack_weights(const float* __restrict__ Wq, const float* __restrict__ Wk,
                  const float* __restrict__ Wv, const float* __restrict__ Wr,
                  _Float16* __restrict__ wpack) {
  const int gid = blockIdx.x * blockDim.x + threadIdx.x;   // 512 frags * 32 lanes
  if (gid >= 4 * Hn * 4 * 4 * 32) return;
  const int lane = gid & 31;
  const int frag = gid >> 5;
  const int kk   = frag & 3;
  const int nt   = (frag >> 2) & 3;
  const int h    = (frag >> 4) & 7;
  const int proj = frag >> 7;
  const float* W = (proj == 0) ? Wq : (proj == 1) ? Wk : (proj == 2) ? Wv : Wr;
  const int half = lane >> 4;
  const int n    = lane & 15;
  v16h t;
#pragma unroll
  for (int v = 0; v < 8; ++v) {
    const int k = kk * 32 + 2 * v + 16 * half;
    t[2 * v]     = (_Float16)W[(k)     * HPn + h * Pn + nt * 16 + n];
    t[2 * v + 1] = (_Float16)W[(k + 1) * HPn + h * Pn + nt * 16 + n];
  }
  *(v16h*)(wpack + (size_t)frag * 512 + lane * 16) = t;
}

// Prep 2: out_w transposed per head: owT[h][p][f] = out_w[f*HP + h*P + p]
__global__ __launch_bounds__(256)
void pack_outw(const float* __restrict__ out_w, float* __restrict__ owT) {
  const int gid = blockIdx.x * blockDim.x + threadIdx.x;
  if (gid < Hn * Pn * Fn) {
    const int f = gid & 63;
    const int p = (gid >> 6) & 63;
    const int h = gid >> 12;
    owT[gid] = out_w[f * HPn + h * Pn + p];
  }
}

// ---------------------------------------------------------------------------
// One block per (batch b, head h). 128 threads = 4 waves.
__global__ __launch_bounds__(128)
void autoint_block(const int*      __restrict__ feat_index,
                   const _Float16* __restrict__ embF16,
                   const _Float16* __restrict__ wpack,
                   const float*    __restrict__ owT,
                   float*          __restrict__ partials) {
  __shared__ _Float16 eS[Fn * Dn];   // 16 KB; reused as f32 scores later (same size)
  __shared__ _Float16 qS[Fn * Pn];   // 8 KB; reused as att (f16)
  __shared__ _Float16 kS[Fn * Pn];   // 8 KB (row-major [f][p])
  __shared__ _Float16 vT[Pn * Fn];   // 8 KB (transposed: [p][f])
  __shared__ _Float16 rT[Pn * Fn];   // 8 KB (residual transposed: [p][f])
  __shared__ float    red[4];

  const int b    = blockIdx.x;
  const int h    = blockIdx.y;
  const int tid  = threadIdx.x;
  const int wave = tid >> 5;
  const int lane = tid & 31;

  // ---- Phase 1: gather e[b] rows (f16, 256B each) via async global->LDS.
  // 1024 16-byte chunks; each of 128 threads issues 8 async b128 copies.
  {
    const uint32_t ldsBase = (uint32_t)(uintptr_t)eS;   // LDS offset (low 32 bits)
#pragma unroll
    for (int i = 0; i < 8; ++i) {
      const int chunk = i * 128 + tid;                  // 0..1023
      const int f     = chunk >> 4;                     // 16 chunks per row
      const int boff  = (chunk & 15) * 16;              // byte offset in row
      const int row   = feat_index[b * Fn + f];
      const uint64_t ga = (uint64_t)(uintptr_t)(embF16 + (size_t)row * Dn) + boff;
      const uint32_t la = ldsBase + (uint32_t)chunk * 16;
      asm volatile("global_load_async_to_lds_b128 %0, %1, off"
                   :: "v"(la), "v"(ga) : "memory");
    }
    asm volatile("s_wait_asynccnt 0x0" ::: "memory");
  }
  __syncthreads();

  // ---- Phase 2: projections. Wave w computes {q,k,v,r}[w] from packed weights.
  {
    const _Float16* wp = wpack + ((size_t)(wave * Hn + h)) * (4 * 4 * 512);
#pragma unroll
    for (int nt = 0; nt < 4; ++nt) {
      v8f acc[4] = {};
#pragma unroll
      for (int kk = 0; kk < 4; ++kk) {
        const v16h bf = *(const v16h*)(wp + (nt * 4 + kk) * 512 + lane * 16);
#pragma unroll
        for (int mt = 0; mt < 4; ++mt) {
          const v16h af = afrag_lds(eS, mt * 16, kk * 32, Dn);
          acc[mt] = wmma_f16(af, bf, acc[mt]);
        }
      }
      if (wave == 0) {
#pragma unroll
        for (int mt = 0; mt < 4; ++mt) cstore_f16(qS, mt * 16, nt * 16, Pn, acc[mt]);
      } else if (wave == 1) {
#pragma unroll
        for (int mt = 0; mt < 4; ++mt) cstore_f16(kS, mt * 16, nt * 16, Pn, acc[mt]);
      } else if (wave == 2) {
#pragma unroll
        for (int mt = 0; mt < 4; ++mt) cstore_f16_T(vT, mt * 16, nt * 16, Fn, acc[mt]);
      } else {
#pragma unroll
        for (int mt = 0; mt < 4; ++mt) cstore_f16_T(rT, mt * 16, nt * 16, Fn, acc[mt]);
      }
    }
  }
  __syncthreads();

  // ---- Phase 3: scores = q @ k^T -> f32 into eS region
  float* scoreS = (float*)eS;  // 64*64 f32 == 16 KB == eS footprint
  {
    const int mt = wave;  // each wave owns one 16-row M strip
#pragma unroll
    for (int nt = 0; nt < 4; ++nt) {
      v8f acc = {};
#pragma unroll
      for (int kk = 0; kk < 2; ++kk) {
        const v16h af = afrag_lds(qS, mt * 16, kk * 32, Pn);
        const v16h bf = bfrag_lds_nk(kS, kk * 32, nt * 16, Pn);  // (K=p,N=fk)=kS[fk][p]
        acc = wmma_f16(af, bf, acc);
      }
      cstore_f32(scoreS, mt * 16, nt * 16, Fn, acc);
    }
  }
  __syncthreads();

  // ---- Phase 4: softmax over the QUERY axis (reference softmax(scores, axis=2)).
  _Float16* attS = qS;   // q is dead
  if (tid < Fn) {
    const int c = tid;
    float mx = -3.0e38f;
    for (int q = 0; q < Fn; ++q) mx = fmaxf(mx, scoreS[q * Fn + c]);
    float s = 0.0f;
    for (int q = 0; q < Fn; ++q) {
      const float e = __expf(scoreS[q * Fn + c] - mx);
      scoreS[q * Fn + c] = e;      // in-place exp
      s += e;
    }
    const float inv = 1.0f / s;
    for (int q = 0; q < Fn; ++q)
      attS[q * Fn + c] = (_Float16)(scoreS[q * Fn + c] * inv);
  }
  __syncthreads();

  // ---- Phase 5: av = att @ v ; fused +residual, ReLU, dot with out_w slice
  float partial = 0.0f;
  {
    const int mt   = wave;
    const int half = lane >> 4;
    const int n    = lane & 15;
    const float* owH = owT + (size_t)h * Pn * Fn;
#pragma unroll
    for (int nt = 0; nt < 4; ++nt) {
      v8f acc = {};
#pragma unroll
      for (int kk = 0; kk < 2; ++kk) {
        const v16h af = afrag_lds(attS, mt * 16, kk * 32, Fn);
        const v16h bf = bfrag_lds_nk(vT, kk * 32, nt * 16, Fn);  // (K=fk,N=p)=vT[p][fk]
        acc = wmma_f16(af, bf, acc);
      }
      const int p  = nt * 16 + n;
      const int fb = mt * 16 + 8 * half;               // first of 8 consecutive f rows
      const v8h rr = *(const v8h*)(rT + p * Fn + fb);  // residual, contiguous
      const v8f ww = *(const v8f*)(owH + p * Fn + fb); // out_w slice, contiguous
#pragma unroll
      for (int v = 0; v < 8; ++v) {
        const float val = fmaxf(acc[v] + (float)rr[v], 0.0f);
        partial += val * ww[v];
      }
    }
  }
  // wave32 reduction, then deterministic per-(b,h) partial (no float atomics)
#pragma unroll
  for (int off = 16; off > 0; off >>= 1)
    partial += __shfl_down(partial, off, 32);
  if (lane == 0) red[wave] = partial;
  __syncthreads();
  if (tid == 0)
    partials[b * Hn + h] = red[0] + red[1] + red[2] + red[3];
}

// Final: y[b] = sigmoid(sum_h partial[b,h] + out_b)
__global__ __launch_bounds__(256)
void autoint_final(const float* __restrict__ partials,
                   const float* __restrict__ out_b,
                   float*       __restrict__ y) {
  const int b = blockIdx.x * blockDim.x + threadIdx.x;
  if (b < Bn) {
    float s = out_b[0];
#pragma unroll
    for (int h = 0; h < Hn; ++h) s += partials[b * Hn + h];
    y[b] = 1.0f / (1.0f + __expf(-s));
  }
}

extern "C" void kernel_launch(void* const* d_in, const int* in_sizes, int n_in,
                              void* d_out, int out_size, void* d_ws, size_t ws_size,
                              hipStream_t stream) {
  (void)in_sizes; (void)n_in; (void)out_size; (void)ws_size;
  const int*   feat_index = (const int*)  d_in[0];
  const float* emb_table  = (const float*)d_in[1];
  const float* Wq         = (const float*)d_in[2];
  const float* Wk         = (const float*)d_in[3];
  const float* Wv         = (const float*)d_in[4];
  const float* Wr         = (const float*)d_in[5];
  const float* out_w      = (const float*)d_in[6];
  const float* out_b      = (const float*)d_in[7];
  float*       y          = (float*)d_out;

  // workspace layout:
  //   [0, 25.6MB)       embF16 (f16 embedding table)
  //   [26.0MB, +512KB)  packed f16 weight fragments
  //   [+512KB, +128KB)  owT
  //   [... , +64KB)     partials
  const size_t OFF_WPACK = 26u * 1024u * 1024u;
  const size_t OFF_OWT   = OFF_WPACK + 512u * 1024u;
  const size_t OFF_PART  = OFF_OWT + 128u * 1024u;
  _Float16* embF16   = (_Float16*)d_ws;
  _Float16* wpack    = (_Float16*)((char*)d_ws + OFF_WPACK);
  float*    owT      = (float*)((char*)d_ws + OFF_OWT);
  float*    partials = (float*)((char*)d_ws + OFF_PART);

  const size_t tblThreads = ((size_t)Vn * Dn + 15) / 16;
  convert_table<<<(unsigned)((tblThreads + 255) / 256), 256, 0, stream>>>(emb_table, embF16);
  pack_weights<<<(4 * Hn * 4 * 4 * 32 + 255) / 256, 256, 0, stream>>>(Wq, Wk, Wv, Wr, wpack);
  pack_outw<<<(Hn * Pn * Fn + 255) / 256, 256, 0, stream>>>(out_w, owT);

  dim3 grid(Bn, Hn);
  autoint_block<<<grid, 128, 0, stream>>>(feat_index, embF16, wpack, owT, partials);
  autoint_final<<<(Bn + 255) / 256, 256, 0, stream>>>(partials, out_b, y);
}
